// MoESpectralHead_48430051230260
// MI455X (gfx1250) — matrix operand order repaired and checked by reading, not compiled, not run on hardware
//
#include <hip/hip_runtime.h>
#include <hip/hip_bf16.h>

// ---------------------------------------------------------------------------
// MoE (router + 8 routed + 2 shared experts), H=HID=OUT=128, 65536 tokens.
// Dense evaluation of all experts (matches reference semantics), bf16 WMMA.
// ---------------------------------------------------------------------------

typedef __attribute__((ext_vector_type(16))) __bf16 v16bf;
typedef __attribute__((ext_vector_type(8)))  __bf16 v8bf;
typedef __attribute__((ext_vector_type(8)))  float  v8f;
typedef __attribute__((ext_vector_type(4)))  float  v4f;
typedef __attribute__((ext_vector_type(16))) float  v16f;

#define NRT 8            // routed experts
#define NEX 10           // routed + shared
#define FR1_OFF 0        // Wr1 fragments: 4 ksteps * 512 = 2048 bf16
#define FW1_OFF 2048     // layer-1 frags: 10 experts * 16384
#define FW2_OFF 165888   // layer-2 frags: 10 experts * 16384
#define FRAG_TOTAL 329728

// ---------------------------------------------------------------------------
// Prep: convert f32 weights -> bf16, pre-swizzled into wave32 B-fragment order.
// B 32x16 bf16 fragment, lane l: N = l&15, K = ((i>>3)<<4) + ((l>>4)<<3) + (i&7), i = 0..15.
// Fragment f = ((t*NT + j)*32 + l)*16 + i ; K = t*32 + Klane, n = j*16 + (l&15)
// ---------------------------------------------------------------------------
__global__ void moe_prep_frags(const float* __restrict__ Wr1,
                               const float* __restrict__ We1,
                               const float* __restrict__ Ws1,
                               const float* __restrict__ We2,
                               const float* __restrict__ Ws2,
                               __bf16* __restrict__ frags) {
    int idx = blockIdx.x * 256 + threadIdx.x;
    if (idx >= FRAG_TOTAL) return;
    if (idx < FW1_OFF) {
        // Wr1: 128x16, NT = 1
        int i = idx & 15, l = (idx >> 4) & 31, t = idx >> 9;
        int K = t * 32 + ((i >> 3) << 4) + ((l >> 4) << 3) + (i & 7);
        int n = l & 15;
        frags[idx] = (__bf16)Wr1[K * 16 + n];
    } else {
        int r = idx - FW1_OFF;
        bool layer2 = (r >= 163840);
        int rr = layer2 ? r - 163840 : r;
        int eg = rr >> 14;          // expert 0..9
        int f  = rr & 16383;
        int i = f & 15, l = (f >> 4) & 31, j = (f >> 9) & 7, t = f >> 12;
        int K = t * 32 + ((i >> 3) << 4) + ((l >> 4) << 3) + (i & 7);
        int n = j * 16 + (l & 15);
        const float* src;
        if (!layer2) src = (eg < NRT) ? We1 + eg * 16384 : Ws1 + (eg - NRT) * 16384;
        else         src = (eg < NRT) ? We2 + eg * 16384 : Ws2 + (eg - NRT) * 16384;
        frags[idx] = (__bf16)src[K * 128 + n];
    }
}

// Whole fragment is contiguous per lane: one 32-byte vector load.
__device__ __forceinline__ v16bf load_bfrag(const __bf16* __restrict__ p, int lane) {
    return *(const v16bf*)(p + lane * 16);
}

// Broadcast a scalar into all 8 accumulator regs (C[m][n] = bias[n]: a lane
// owns a fixed column n in the C/D layout, so this seeds the bias add for free).
__device__ __forceinline__ v8f splat8(float v) {
    v8f r;
#pragma unroll
    for (int i = 0; i < 8; ++i) r[i] = v;
    return r;
}

// gelu_tanh(v) = 0.5*v*(1+tanh(u)) = v * sigmoid(2u),  u = c*(v + 0.044715 v^3)
// sigmoid(2u) = rcp(1 + exp2(-2*c*log2(e) * w)),  2*0.79788456*1.44269504 = 2.3022077
__device__ __forceinline__ float gelu_tanh(float v) {
    float w = __builtin_fmaf(0.044715f * v, v * v, v);
    float e = __builtin_amdgcn_exp2f(w * -2.3022077f);
    return v * __builtin_amdgcn_rcpf(1.0f + e);
}

// ---------------------------------------------------------------------------
// Main: 256 threads = 8 waves per WG; one 16-token tile per wave.
// ---------------------------------------------------------------------------
__global__ void __launch_bounds__(256)
moe_main(const float* __restrict__ x,
         const float* __restrict__ Wr2, const float* __restrict__ br1,
         const float* __restrict__ br2,
         const float* __restrict__ be1, const float* __restrict__ be2,
         const float* __restrict__ bs1, const float* __restrict__ bs2,
         const __bf16* __restrict__ frags,
         float* __restrict__ out) {
    __shared__ __bf16 Hl[8][16][128];   // per-wave hidden tile (bf16)
    __shared__ float  rfl[8][16][16];   // per-wave router features
    __shared__ float  gl[8][16][8];     // per-wave gates

    const int lane = threadIdx.x & 31;
    const int wv   = threadIdx.x >> 5;
    const int tile = blockIdx.x * 8 + wv;
    const int tok0 = tile * 16;
    const int mrow = lane & 15;     // A-layout row / C-layout column
    const int half = lane >> 4;

    // ---- load X tile once as 4 bf16 A-fragments (16x32 each), kept in VGPRs
    // A layout: lane holds M = lane&15; i=0..7 -> K = t*32 + half*8 + i,
    //                                  i=8..15 -> K = t*32 + 16 + half*8 + (i-8)
    v16bf a[4];
    {
        const float* xrow = x + (size_t)(tok0 + mrow) * 128;
#pragma unroll
        for (int t = 0; t < 4; ++t) {
            const int k0 = t * 32 + half * 8;
            v4f f0 = *(const v4f*)(xrow + k0);
            v4f f1 = *(const v4f*)(xrow + k0 + 4);
            v4f f2 = *(const v4f*)(xrow + k0 + 16);
            v4f f3 = *(const v4f*)(xrow + k0 + 20);
            v16f tmp;
#pragma unroll
            for (int i = 0; i < 4; ++i) {
                tmp[i]      = f0[i];
                tmp[4 + i]  = f1[i];
                tmp[8 + i]  = f2[i];
                tmp[12 + i] = f3[i];
            }
            a[t] = __builtin_convertvector(tmp, v16bf);
        }
    }

    // ---- router: rf = relu(X @ Wr1 + br1) via WMMA (N-tile = 16 = RDIM)
    {
        v8f rf = splat8(br1[mrow]);           // bias seeded into accumulator
#pragma unroll
        for (int t = 0; t < 4; ++t)
            rf = __builtin_amdgcn_wmma_f32_16x16x32_bf16(
                false, a[t], false, load_bfrag(frags + FR1_OFF + t * 512, lane),
                (short)0, rf, false, false);
#pragma unroll
        for (int r = 0; r < 8; ++r)
            rfl[wv][half * 8 + r][mrow] = rf[r] > 0.0f ? rf[r] : 0.0f;
    }
    // ---- logits + softmax + threshold gate (f32 VALU, token m = lane&15)
    {
        float lg[8];
#pragma unroll
        for (int e = 0; e < 8; ++e) lg[e] = br2[e];
#pragma unroll
        for (int j = 0; j < 16; ++j) {
            float rv = rfl[wv][mrow][j];
#pragma unroll
            for (int e = 0; e < 8; ++e) lg[e] += rv * Wr2[j * 8 + e];
        }
        float mx = lg[0];
#pragma unroll
        for (int e = 1; e < 8; ++e) mx = fmaxf(mx, lg[e]);
        float s = 0.0f;
#pragma unroll
        for (int e = 0; e < 8; ++e) {
            lg[e] = __builtin_amdgcn_exp2f((lg[e] - mx) * 1.44269504f);
            s += lg[e];
        }
        float inv = __builtin_amdgcn_rcpf(s);
        float msum = 0.0f;
#pragma unroll
        for (int e = 0; e < 8; ++e) {
            float g = lg[e] * inv;
            g = (g > 0.125f) ? g : 0.0f;   // threshold = 1/NR
            lg[e] = g; msum += g;
        }
        float rinv = __builtin_amdgcn_rcpf(msum + 1e-8f);
        if (lane < 16) {
#pragma unroll
            for (int e = 0; e < 8; ++e) gl[wv][mrow][e] = lg[e] * rinv;
        }
    }

    // ---- dense expert evaluation, gate-weighted accumulation
    float oacc[8][8];                    // [ntile j][r]
#pragma unroll
    for (int j = 0; j < 8; ++j)
#pragma unroll
        for (int r = 0; r < 8; ++r) oacc[j][r] = 0.0f;

#pragma unroll 1
    for (int e = 0; e < NEX; ++e) {
        const __bf16* W1 = frags + FW1_OFF + e * 16384;
        const __bf16* W2 = frags + FW2_OFF + e * 16384;
        const float* bias1 = (e < NRT) ? be1 + e * 128 : bs1 + (e - NRT) * 128;
        const float* bias2 = (e < NRT) ? be2 + e * 128 : bs2 + (e - NRT) * 128;

        // layer 1: H = gelu(X @ W1 + b1) -> bf16 LDS tile (bias in accumulator)
#pragma unroll
        for (int j = 0; j < 8; ++j) {
            v8f acc = splat8(bias1[j * 16 + mrow]);
#pragma unroll
            for (int t = 0; t < 4; ++t)
                acc = __builtin_amdgcn_wmma_f32_16x16x32_bf16(
                    false, a[t], false, load_bfrag(W1 + (t * 8 + j) * 512, lane),
                    (short)0, acc, false, false);
#pragma unroll
            for (int r = 0; r < 8; ++r)
                Hl[wv][half * 8 + r][j * 16 + mrow] = (__bf16)gelu_tanh(acc[r]);
        }

        // reload H as A-fragments (two 16B LDS vector loads per fragment)
        v16bf h[4];
#pragma unroll
        for (int t = 0; t < 4; ++t) {
            const int k0 = t * 32 + half * 8;
            union { struct { v8bf lo, hi; } s; v16bf v; } u;
            u.s.lo = *(const v8bf*)&Hl[wv][mrow][k0];
            u.s.hi = *(const v8bf*)&Hl[wv][mrow][k0 + 16];
            h[t] = u.v;
        }

        // gate weights for this lane's 8 output rows
        float gw[8];
        if (e < NRT) {
#pragma unroll
            for (int r = 0; r < 8; ++r) gw[r] = gl[wv][half * 8 + r][e];
        } else {
#pragma unroll
            for (int r = 0; r < 8; ++r) gw[r] = 1.0f;
        }

        // layer 2: O = H @ W2 + b2 (bias in accumulator), weighted accumulate
#pragma unroll
        for (int j = 0; j < 8; ++j) {
            v8f acc = splat8(bias2[j * 16 + mrow]);
#pragma unroll
            for (int t = 0; t < 4; ++t)
                acc = __builtin_amdgcn_wmma_f32_16x16x32_bf16(
                    false, h[t], false, load_bfrag(W2 + (t * 8 + j) * 512, lane),
                    (short)0, acc, false, false);
#pragma unroll
            for (int r = 0; r < 8; ++r)
                oacc[j][r] = __builtin_fmaf(gw[r], acc[r], oacc[j][r]);
        }
    }

    // ---- store: C/D layout -> out[token][n], token = tok0 + half*8 + r
#pragma unroll
    for (int r = 0; r < 8; ++r) {
        float* orow = out + (size_t)(tok0 + half * 8 + r) * 128 + mrow;
#pragma unroll
        for (int j = 0; j < 8; ++j) orow[j * 16] = oacc[j][r];
    }
}

extern "C" void kernel_launch(void* const* d_in, const int* in_sizes, int n_in,
                              void* d_out, int out_size, void* d_ws, size_t ws_size,
                              hipStream_t stream) {
    const float* x   = (const float*)d_in[0];
    const float* Wr1 = (const float*)d_in[1];
    const float* br1 = (const float*)d_in[2];
    const float* Wr2 = (const float*)d_in[3];
    const float* br2 = (const float*)d_in[4];
    const float* We1 = (const float*)d_in[5];
    const float* be1 = (const float*)d_in[6];
    const float* We2 = (const float*)d_in[7];
    const float* be2 = (const float*)d_in[8];
    const float* Ws1 = (const float*)d_in[9];
    const float* bs1 = (const float*)d_in[10];
    const float* Ws2 = (const float*)d_in[11];
    const float* bs2 = (const float*)d_in[12];
    __bf16* frags = (__bf16*)d_ws;            // ~660 KB of scratch used
    float* out = (float*)d_out;

    int ntok  = in_sizes[0] / 128;            // 65536
    int tiles = ntok / 16;                    // 4096

    moe_prep_frags<<<(FRAG_TOTAL + 255) / 256, 256, 0, stream>>>(
        Wr1, We1, Ws1, We2, Ws2, frags);
    moe_main<<<tiles / 8, 256, 0, stream>>>(
        x, Wr2, br1, br2, be1, be2, bs1, bs2, frags, out);
}
